// ParallelGWAttention_46669114638477
// MI455X (gfx1250) — matrix operand order repaired and checked by reading, not compiled, not run on hardware
//
#include <hip/hip_runtime.h>
#include <hip/hip_bf16.h>

typedef __attribute__((ext_vector_type(16))) _Float16 v16h;
typedef __attribute__((ext_vector_type(8)))  _Float16 v8h;
typedef __attribute__((ext_vector_type(8)))  float    v8f;

#define WMMA(a, b, c) __builtin_amdgcn_wmma_f32_16x16x32_f16(false, (a), false, (b), (short)0, (c), false, false)

// ---- geometry ----
constexpr int TOK    = 64;    // tokens per 8x8 window
constexpr int CH     = 256;   // channels

// ---- LDS layout (bytes) ----
constexpr int XS_STR  = 272;   // f16 row stride for X   [64][272]
constexpr int QKV_STR = 1040;  // f16 row stride for QKV [64][1040]
constexpr int KVT_STR = 72;    // f16 row stride for kv^T [16][32][72]
constexpr int G_STR   = 20;    // f32 row stride for gate [64][20]
constexpr int O_STR   = 272;   // f16 row stride for O   [2][64][272]
constexpr int Y_STR   = 264;   // f32 row stride for staged Y (overlays QKVS)

constexpr int OFF_XS   = 0;
constexpr int OFF_QKVS = OFF_XS   + TOK * XS_STR  * 2;     // 34816
constexpr int OFF_KVT  = OFF_QKVS + TOK * QKV_STR * 2;     // 167936
constexpr int OFF_G    = OFF_KVT  + 16 * 32 * KVT_STR * 2; // 241664
constexpr int OFF_O    = OFF_G    + TOK * G_STR * 4;       // 246784
constexpr int SMEM_BYTES = OFF_O + 2 * TOK * O_STR * 2;    // 316416 (<= 320KB WGP)

// ---- weight sizes (elements) ----
constexpr int WQ_N = 1024 * 256;
constexpr int WG_N = 16 * 256;
constexpr int WO_N = 512 * 256;

// A fragment: 16-bit A 16x32 layout. lanes 0-15 -> M=r, K = k0+{0..7,16..23};
// lanes 16-31 -> M=r, K = k0+{8..15,24..31}.
__device__ __forceinline__ v16h load_a_frag(const _Float16* base, int stride, int row0, int k0) {
    const int lane = threadIdx.x & 31;
    const int r = lane & 15, half = lane >> 4;
    const _Float16* p = base + (row0 + r) * stride + k0 + half * 8;
    v16h a;
    ((v8h*)&a)[0] = *(const v8h*)(p);
    ((v8h*)&a)[1] = *(const v8h*)(p + 16);
    return a;
}

// B fragment: 16-bit B 32x16 layout. lanes 0-15 -> N=r, K = k0+{0..7,16..23};
// lanes 16-31 -> N=r, K = k0+{8..15,24..31}. Source matrix is row-major [N][K] ("B^T rows").
__device__ __forceinline__ v16h load_b_frag(const _Float16* base, int stride, int col0, int k0) {
    const int lane = threadIdx.x & 31;
    const int r = lane & 15, half = lane >> 4;
    const _Float16* p = base + (col0 + r) * stride + k0 + half * 8;
    v16h b;
    ((v8h*)&b)[0] = *(const v8h*)(p);
    ((v8h*)&b)[1] = *(const v8h*)(p + 16);
    return b;
}

// ---- prep: convert f32 weights to f16 in workspace ----
__global__ void weights_to_f16_kernel(const float* __restrict__ Wq,
                                      const float* __restrict__ Wg,
                                      const float* __restrict__ Wo,
                                      _Float16* __restrict__ ws) {
    int i = blockIdx.x * blockDim.x + threadIdx.x;
    if (i < WQ_N) {
        ws[i] = (_Float16)Wq[i];
    } else if (i < WQ_N + WG_N) {
        ws[i] = (_Float16)Wg[i - WQ_N];
    } else if (i < WQ_N + WG_N + WO_N) {
        ws[i] = (_Float16)Wo[i - WQ_N - WG_N];
    }
}

__global__ __launch_bounds__(256, 1)
void gw_attention_kernel(const float* __restrict__ x,
                         const _Float16* __restrict__ Wq16,   // [1024][256]
                         const _Float16* __restrict__ Wg16,   // [16][256]
                         const _Float16* __restrict__ Wo16,   // [512][256]
                         const float* __restrict__ b_gate,    // [16]
                         float* __restrict__ out) {
    extern __shared__ char smem[];
    _Float16* XS   = (_Float16*)(smem + OFF_XS);
    _Float16* QKVS = (_Float16*)(smem + OFF_QKVS);
    _Float16* KVT  = (_Float16*)(smem + OFF_KVT);
    float*    G    = (float*)   (smem + OFF_G);
    _Float16* O    = (_Float16*)(smem + OFF_O);
    float*    YST  = (float*)   (smem + OFF_QKVS);  // overlays QKVS (free in phase E)

    const int tid  = threadIdx.x;
    const int wv   = tid >> 5;
    const int lane = tid & 31;
    const int cl   = lane & 15;
    const int half = lane >> 4;

    // window decode
    const int w   = blockIdx.x;
    const int b0  = w >> 8;
    const int rem = w & 255;
    const int h0  = (rem >> 4) * 8;
    const int w0  = (rem & 15) * 8;
    const long base_img = ((long)b0 * CH) * 128 * 128;

    // ---------------- Phase A: gather window -> XS (f16), float4 loads ----------------
    // 4096 float4 = 64 tok * 256 ch; per channel: 8 rows (p1) x 2 float4 (p2)
    #pragma unroll 4
    for (int i = 0; i < 16; ++i) {
        int v    = i * 256 + tid;
        int c    = v >> 4;
        int rv   = v & 15;
        int p1   = rv >> 1;
        int p2b  = (rv & 1) * 4;
        const float4 f = *(const float4*)(x + base_img + ((long)c * 128 + (h0 + p1)) * 128 + (w0 + p2b));
        int tokb = p1 * 8 + p2b;
        XS[(tokb + 0) * XS_STR + c] = (_Float16)f.x;
        XS[(tokb + 1) * XS_STR + c] = (_Float16)f.y;
        XS[(tokb + 2) * XS_STR + c] = (_Float16)f.z;
        XS[(tokb + 3) * XS_STR + c] = (_Float16)f.w;
    }
    __syncthreads();

    // ---------------- Phase B: QKV GEMM (+ gate, wave 0) ----------------
    // qkv[t][c] = sum_k XS[t][k] * Wq[c][k];  64 N-tiles over 8 waves
    for (int nt = wv; nt < 64; nt += 8) {
        const int col0 = nt * 16;
        v8f acc[4];
        #pragma unroll
        for (int mi = 0; mi < 4; ++mi) acc[mi] = (v8f){0.f,0.f,0.f,0.f,0.f,0.f,0.f,0.f};
        for (int k0 = 0; k0 < 256; k0 += 32) {
            v16h bf = load_b_frag(Wq16, 256, col0, k0);
            #pragma unroll
            for (int mi = 0; mi < 4; ++mi) {
                v16h af = load_a_frag(XS, XS_STR, mi * 16, k0);
                acc[mi] = WMMA(af, bf, acc[mi]);
            }
        }
        const bool is_kv = (col0 >= 512);
        const int rel = col0 + cl - 512;           // only used when is_kv
        const int pr = rel >> 5, dd = rel & 31;    // pr = p*8+h, dd = d
        #pragma unroll
        for (int mi = 0; mi < 4; ++mi) {
            #pragma unroll
            for (int r = 0; r < 8; ++r) {
                int t = mi * 16 + r + half * 8;
                _Float16 hv = (_Float16)acc[mi][r];
                QKVS[t * QKV_STR + col0 + cl] = hv;
                if (is_kv) KVT[(pr * 32 + dd) * KVT_STR + t] = hv;
            }
        }
    }
    if (wv == 0) {
        // gate = XS @ Wg^T + b  (one 16-wide N tile)
        v8f gacc[4];
        #pragma unroll
        for (int mi = 0; mi < 4; ++mi) gacc[mi] = (v8f){0.f,0.f,0.f,0.f,0.f,0.f,0.f,0.f};
        for (int k0 = 0; k0 < 256; k0 += 32) {
            v16h bf = load_b_frag(Wg16, 256, 0, k0);
            #pragma unroll
            for (int mi = 0; mi < 4; ++mi) {
                v16h af = load_a_frag(XS, XS_STR, mi * 16, k0);
                gacc[mi] = WMMA(af, bf, gacc[mi]);
            }
        }
        float bias = b_gate[cl];
        #pragma unroll
        for (int mi = 0; mi < 4; ++mi) {
            #pragma unroll
            for (int r = 0; r < 8; ++r) {
                int t = mi * 16 + r + half * 8;
                float v = gacc[mi][r] + bias;
                G[t * G_STR + cl] = 1.0f / (1.0f + __expf(-v));   // sigmoid
            }
        }
    }
    __syncthreads();

    // ---------------- Phase C: attention, 2 (p,h) problems per wave ----------------
    const float scale = 0.17677669529663687f;  // 1/sqrt(32)
    for (int pi = 0; pi < 2; ++pi) {
        const int prob = wv + pi * 8;          // 0..15
        const int p = prob >> 3, h = prob & 7;
        const int qc0 = p * 256 + h * 32;
        const int kc0 = 512 + qc0;
        const int pr  = prob;

        // S^T = kv @ q^T  -> st[n_tile][m_tile], D layout: lane = n, vgpr = m
        v8f st[4][4];
        #pragma unroll
        for (int ni = 0; ni < 4; ++ni) {
            #pragma unroll
            for (int mi = 0; mi < 4; ++mi) {
                v16h af = load_a_frag(QKVS, QKV_STR, mi * 16, kc0);   // kv rows (K = d, 32)
                v16h bf = load_b_frag(QKVS, QKV_STR, ni * 16, qc0);   // q rows as B columns
                v8f z = (v8f){0.f,0.f,0.f,0.f,0.f,0.f,0.f,0.f};
                st[ni][mi] = WMMA(af, bf, z);
            }
        }
        // softmax over m: per lane, values split only across lane^16
        #pragma unroll
        for (int ni = 0; ni < 4; ++ni) {
            float mx = -3.4e38f;
            #pragma unroll
            for (int mi = 0; mi < 4; ++mi)
                #pragma unroll
                for (int r = 0; r < 8; ++r) mx = fmaxf(mx, st[ni][mi][r]);
            mx = fmaxf(mx, __shfl_xor(mx, 16, 32));
            float s = 0.f;
            #pragma unroll
            for (int mi = 0; mi < 4; ++mi)
                #pragma unroll
                for (int r = 0; r < 8; ++r) {
                    float e = __expf((st[ni][mi][r] - mx) * scale);
                    st[ni][mi][r] = e;
                    s += e;
                }
            s += __shfl_xor(s, 16, 32);
            float rinv = 1.0f / s;
            #pragma unroll
            for (int mi = 0; mi < 4; ++mi)
                #pragma unroll
                for (int r = 0; r < 8; ++r) st[ni][mi][r] *= rinv;
        }
        // o = attn @ kv : A comes straight from st registers (layout-exact), B from KVT
        #pragma unroll
        for (int ti = 0; ti < 4; ++ti) {
            v8f acc0 = (v8f){0.f,0.f,0.f,0.f,0.f,0.f,0.f,0.f};
            v8f acc1 = acc0;
            #pragma unroll
            for (int kk = 0; kk < 2; ++kk) {
                v16h af;
                #pragma unroll
                for (int e = 0; e < 16; ++e)
                    af[e] = (_Float16)st[ti][2 * kk + (e >> 3)][e & 7];
                v16h b0 = load_b_frag(KVT, KVT_STR, pr * 32 + 0,  kk * 32);
                v16h b1 = load_b_frag(KVT, KVT_STR, pr * 32 + 16, kk * 32);
                acc0 = WMMA(af, b0, acc0);
                acc1 = WMMA(af, b1, acc1);
            }
            // gate + store O[p][t][h*32 + d]
            #pragma unroll
            for (int r = 0; r < 8; ++r) {
                int t = ti * 16 + r + half * 8;
                float g = G[t * G_STR + pr];
                O[(p * TOK + t) * O_STR + h * 32 + 0  + cl] = (_Float16)(acc0[r] * g);
                O[(p * TOK + t) * O_STR + h * 32 + 16 + cl] = (_Float16)(acc1[r] * g);
            }
        }
    }
    __syncthreads();

    // ---------------- Phase E: output projection, sum both branches ----------------
    // y[t][c] = sum_p sum_k O[p][t][k] * Wo[p*256 + c][k]; 16 N-tiles over 8 waves
    for (int pi = 0; pi < 2; ++pi) {
        const int ct = wv + pi * 8;            // 0..15
        v8f acc[4];
        #pragma unroll
        for (int mi = 0; mi < 4; ++mi) acc[mi] = (v8f){0.f,0.f,0.f,0.f,0.f,0.f,0.f,0.f};
        #pragma unroll
        for (int p = 0; p < 2; ++p) {
            for (int k0 = 0; k0 < 256; k0 += 32) {
                v16h bf = load_b_frag(Wo16, 256, p * 256 + ct * 16, k0);
                #pragma unroll
                for (int mi = 0; mi < 4; ++mi) {
                    v16h af = load_a_frag(O, O_STR, p * TOK + mi * 16, k0);
                    acc[mi] = WMMA(af, bf, acc[mi]);
                }
            }
        }
        #pragma unroll
        for (int mi = 0; mi < 4; ++mi)
            #pragma unroll
            for (int r = 0; r < 8; ++r) {
                int t = mi * 16 + r + half * 8;
                YST[t * Y_STR + ct * 16 + cl] = acc[mi][r];
            }
    }
    __syncthreads();

    // ---------------- write-out, coalesced float4 along W ----------------
    #pragma unroll 4
    for (int i = 0; i < 16; ++i) {
        int v    = i * 256 + tid;
        int c    = v >> 4;
        int rv   = v & 15;
        int p1   = rv >> 1;
        int p2b  = (rv & 1) * 4;
        int tokb = p1 * 8 + p2b;
        float4 f;
        f.x = YST[(tokb + 0) * Y_STR + c];
        f.y = YST[(tokb + 1) * Y_STR + c];
        f.z = YST[(tokb + 2) * Y_STR + c];
        f.w = YST[(tokb + 3) * Y_STR + c];
        *(float4*)(out + base_img + ((long)c * 128 + (h0 + p1)) * 128 + (w0 + p2b)) = f;
    }
}

extern "C" void kernel_launch(void* const* d_in, const int* in_sizes, int n_in,
                              void* d_out, int out_size, void* d_ws, size_t ws_size,
                              hipStream_t stream) {
    const float* x      = (const float*)d_in[0];
    const float* W_qkv  = (const float*)d_in[1];
    const float* W_gate = (const float*)d_in[2];
    const float* b_gate = (const float*)d_in[3];
    const float* W_out  = (const float*)d_in[4];
    float* out = (float*)d_out;

    _Float16* ws16 = (_Float16*)d_ws;
    const _Float16* Wq16 = ws16;
    const _Float16* Wg16 = ws16 + WQ_N;
    const _Float16* Wo16 = ws16 + WQ_N + WG_N;

    const int total_w = WQ_N + WG_N + WO_N;
    weights_to_f16_kernel<<<(total_w + 255) / 256, 256, 0, stream>>>(W_qkv, W_gate, W_out, ws16);

    // 2048 windows: 8 batches * 16 * 16
    gw_attention_kernel<<<2048, 256, SMEM_BYTES, stream>>>(x, Wq16, Wg16, Wo16, b_gate, out);
}
